// BlockSparseSelfAttention_40235253629091
// MI455X (gfx1250) — matrix operand order repaired and checked by reading, not compiled, not run on hardware
//
#include <hip/hip_runtime.h>
#include <hip/hip_bf16.h>

typedef __attribute__((ext_vector_type(16))) __bf16 v16bf;
typedef __attribute__((ext_vector_type(8)))  float  v8f;

#define BATCH 32
#define SEQ   4000
#define DIM   256
#define BS    20
#define NB    200   // SEQ / BS
#define D3    768   // 3*DIM
#define NQKV  (D3 * DIM)    // 196608
#define NOUT  (DIM * DIM)   // 65536

static __device__ __forceinline__ unsigned short f2bf(float f) {
    unsigned int u = __float_as_uint(f);
    unsigned int r = u + 0x7FFFu + ((u >> 16) & 1u);   // round-to-nearest-even
    return (unsigned short)(r >> 16);
}
static __device__ __forceinline__ float bf2f(unsigned short h) {
    return __uint_as_float(((unsigned int)h) << 16);
}

union U16 { v16bf v; uint4 q[2]; };

// A fragment (16x32 bf16): lane L -> row m0 + (L&15); khalf = L>>4.
// elems 0-7 : cols k0 + khalf*8 + e        (16 contiguous bytes)
// elems 8-15: cols k0 + 16 + khalf*8 + e   (16 contiguous bytes)
static __device__ __forceinline__ v16bf load_a(const unsigned short* base, int stride,
                                               int m0, int k0, int lm, int lh) {
    const unsigned short* p = base + (m0 + lm) * stride + k0 + lh * 8;
    U16 u;
    u.q[0] = *(const uint4*)(p);
    u.q[1] = *(const uint4*)(p + 16);
    return u.v;
}

// B fragment (32x16 bf16, N-major per lane): lane L -> col n0 + (L&15); khalf = L>>4.
// elems 0-7 : k = k0 + khalf*8 + e; elems 8-15: k = k0 + 16 + khalf*8 + (e-8)
static __device__ __forceinline__ v16bf load_b(const unsigned short* base, int stride,
                                               int n0, int k0, int lm, int lh) {
    const unsigned short* p = base + (n0 + lm) * stride + k0 + lh * 8;
    U16 u;
    u.q[0] = *(const uint4*)(p);
    u.q[1] = *(const uint4*)(p + 16);
    return u.v;
}

static __device__ __forceinline__ v8f wmma_bf16(v16bf a, v16bf b, v8f c) {
    return __builtin_amdgcn_wmma_f32_16x16x32_bf16(
        /*neg_a=*/false, a, /*neg_b=*/false, b,
        /*c_mod=*/(short)0, c, /*reuse_a=*/false, /*reuse_b=*/false);
}

// ---------------------------------------------------------------------------
// Weight fp32 -> bf16 conversion (weights then live in L2 for all 6400 WGs)
// ---------------------------------------------------------------------------
__global__ void cvt_weights(const float* __restrict__ wq, const float* __restrict__ wo,
                            unsigned short* __restrict__ dq, unsigned short* __restrict__ dw) {
    int i = blockIdx.x * blockDim.x + threadIdx.x;
    if (i < NQKV)             dq[i]         = f2bf(wq[i]);
    else if (i < NQKV + NOUT) dw[i - NQKV]  = f2bf(wo[i - NQKV]);
}

// ---------------------------------------------------------------------------
// One workgroup = one (batch, block). 256 threads = 8 waves.
// ---------------------------------------------------------------------------
__global__ __launch_bounds__(256) void block_attn_kernel(
        const float* __restrict__ x,
        const float* __restrict__ in_b,
        const float* __restrict__ out_b,
        const unsigned short* __restrict__ wqkv,  // [768][256] bf16
        const unsigned short* __restrict__ wout,  // [256][256] bf16
        float* __restrict__ out) {

    // LDS regions (total 64 KB), phase-overlaid:
    __shared__ __align__(16) unsigned short s_qk[32 * 512];  // 32KB: Q|K (ph1-2) -> O (ph4-5)
    __shared__ __align__(16) unsigned short s_xb[32 * 256];  // 16KB: Xb (ph1)  -> scores+attn (ph2-4)
    __shared__ __align__(16) unsigned short s_vT[256 * 32];  // 16KB: V transposed [feat][row]

    unsigned short* s_o  = s_qk;            // 32 x 256 bf16
    unsigned short* s_sc = s_xb;            // 4 x 32 x 32 bf16 scores
    unsigned short* s_at = s_xb + 4096;     // 4 x 32 x 32 bf16 attn

    const int tid  = threadIdx.x;
    const int w    = tid >> 5;
    const int lane = tid & 31;
    const int lm   = lane & 15;
    const int lh   = lane >> 4;

    const int bi  = blockIdx.x / NB;
    const int blk = blockIdx.x % NB;
    const float* xrow = x + ((size_t)bi * SEQ + (size_t)blk * BS) * DIM;

    // ---- Phase 0: stage x block into LDS as bf16, rows 20..31 zero ----
    for (int i = tid; i < 32 * 256; i += 256) {
        int r = i >> 8, c = i & 255;
        s_xb[i] = (r < BS) ? f2bf(xrow[r * DIM + c]) : (unsigned short)0;
    }
    __syncthreads();

    // ---- Phase 1: QKV = Xb (32x256) @ Wqkv^T (256x768), bias add ----
    // 2 M-tiles x 48 N-tiles = 96 tiles, 8 K-steps each
    for (int t = w; t < 96; t += 8) {
        const int mt = t & 1, nt = t >> 1;
        const int m0 = mt * 16, n0 = nt * 16;
        v8f c = {};
        for (int k0 = 0; k0 < 256; k0 += 32) {
            v16bf a = load_a(s_xb, 256, m0, k0, lm, lh);
            v16bf b = load_b(wqkv, 256, n0, k0, lm, lh);
            c = wmma_bf16(a, b, c);
        }
        const float bias = in_b[n0 + lm];
        if (n0 < 512) {                       // Q or K -> row-major
            for (int r = 0; r < 8; ++r) {
                int row = m0 + lh * 8 + r;
                s_qk[row * 512 + n0 + lm] = f2bf(c[r] + bias);
            }
        } else {                              // V -> transposed [feat][row]
            const int f = n0 - 512 + lm;
            for (int r = 0; r < 8; ++r) {
                int row = m0 + lh * 8 + r;
                s_vT[f * 32 + row] = f2bf(c[r] + bias);
            }
        }
    }
    __syncthreads();

    // ---- Phase 2: scores = (Q*scale) @ K^T per head ----
    // 4 heads x 2 Mtiles x 2 Ntiles = 16 tiles, K=64 (2 steps)
    for (int t = w; t < 16; t += 8) {
        const int h = t >> 2, mt = (t >> 1) & 1, nt = t & 1;
        const int m0 = mt * 16;
        v8f c = {};
        for (int k0 = 0; k0 < 64; k0 += 32) {
            v16bf a = load_a(s_qk,       512, m0,      h * 64 + k0, lm, lh);
            v16bf b = load_b(s_qk + 256, 512, nt * 16, h * 64 + k0, lm, lh);
            c = wmma_bf16(a, b, c);
        }
        for (int r = 0; r < 8; ++r) {
            int row = m0 + lh * 8 + r;
            s_sc[h * 1024 + row * 32 + nt * 16 + lm] = f2bf(c[r] * 0.125f);
        }
    }
    __syncthreads();

    // ---- Phase 3: softmax over the 20 valid keys (thread per head-row) ----
    if (tid < 128) {
        const int h = tid >> 5, row = tid & 31;
        const unsigned short* sr = s_sc + h * 1024 + row * 32;
        float vals[BS];
        float mx = -1e30f;
        for (int j = 0; j < BS; ++j) { float v = bf2f(sr[j]); vals[j] = v; mx = fmaxf(mx, v); }
        float sum = 0.f;
        for (int j = 0; j < BS; ++j) { float e = __expf(vals[j] - mx); vals[j] = e; sum += e; }
        const float inv = 1.f / sum;
        unsigned short* ar = s_at + h * 1024 + row * 32;
        for (int j = 0; j < BS; ++j) ar[j] = f2bf(vals[j] * inv);
        for (int j = BS; j < 32; ++j) ar[j] = 0;   // padded keys contribute 0
    }
    __syncthreads();

    // ---- Phase 4: O = attn (32x32) @ V (32x64) per head ----
    // 4 heads x 2 Mtiles x 4 Etiles = 32 tiles, single K=32 step
    for (int t = w; t < 32; t += 8) {
        const int h = t >> 3, mt = (t >> 2) & 1, et = t & 3;
        const int m0 = mt * 16;
        const int f  = h * 64 + et * 16;
        v8f c = {};
        v16bf a = load_a(s_at + h * 1024, 32, m0, 0, lm, lh);
        v16bf b = load_b(s_vT,            32, f,  0, lm, lh);
        c = wmma_bf16(a, b, c);
        for (int r = 0; r < 8; ++r) {
            int row = m0 + lh * 8 + r;
            s_o[row * 256 + f + lm] = f2bf(c[r]);   // overlays dead s_qk
        }
    }
    __syncthreads();

    // ---- Phase 5: final = O (32x256) @ Wout^T (256x256) + bias -> global ----
    // 2 Mtiles x 16 Ntiles = 32 tiles, 8 K-steps
    float* orow = out + ((size_t)bi * SEQ + (size_t)blk * BS) * DIM;
    for (int t = w; t < 32; t += 8) {
        const int mt = t & 1, nt = t >> 1;
        const int m0 = mt * 16, n0 = nt * 16;
        v8f c = {};
        for (int k0 = 0; k0 < 256; k0 += 32) {
            v16bf a = load_a(s_o,  256, m0, k0, lm, lh);
            v16bf b = load_b(wout, 256, n0, k0, lm, lh);
            c = wmma_bf16(a, b, c);
        }
        const float bias = out_b[n0 + lm];
        for (int r = 0; r < 8; ++r) {
            int s = m0 + lh * 8 + r;
            if (s < BS) orow[(size_t)s * DIM + n0 + lm] = c[r] + bias;
        }
    }
}

extern "C" void kernel_launch(void* const* d_in, const int* in_sizes, int n_in,
                              void* d_out, int out_size, void* d_ws, size_t ws_size,
                              hipStream_t stream) {
    const float* x     = (const float*)d_in[0];
    const float* wq_f  = (const float*)d_in[1];   // in_proj_w  [768,256]
    const float* inb   = (const float*)d_in[2];   // in_proj_b  [768]
    const float* wo_f  = (const float*)d_in[3];   // out_proj_w [256,256]
    const float* outb  = (const float*)d_in[4];   // out_proj_b [256]
    float* out = (float*)d_out;

    unsigned short* wqkv_bf = (unsigned short*)d_ws;           // 196608 elems
    unsigned short* wout_bf = wqkv_bf + NQKV;                  // 65536 elems

    {
        const int n = NQKV + NOUT;
        cvt_weights<<<(n + 255) / 256, 256, 0, stream>>>(wq_f, wo_f, wqkv_bf, wout_bf);
    }

    block_attn_kernel<<<BATCH * NB, 256, 0, stream>>>(x, inb, outb, wqkv_bf, wout_bf, out);
}
